// HierarchicalPCN_61108794688125
// MI455X (gfx1250) — matrix operand order repaired and checked by reading, not compiled, not run on hardware
//
#include <hip/hip_runtime.h>
#include <cstdint>
#include <cstddef>

// ---------------------------------------------------------------------------
// HierarchicalPCN on MI455X (gfx1250): bf16 WMMA GEMMs with fused PC epilogues
// B tiles staged with GLOBAL_LOAD_ASYNC_TO_LDS_B128 (ASYNCcnt-tracked DMA)
// ---------------------------------------------------------------------------

#define DT_C   0.01f
#define LAMB_C 0.001f

typedef __attribute__((ext_vector_type(16))) __bf16 v16bf;
typedef __attribute__((ext_vector_type(8)))  float  v8f;

union V16U { v16bf v; uint4 u[2]; };

constexpr int Bsz = 4096;   // batch
constexpr int Dm  = 2048;   // hidden/sensory dim
constexpr int D0  = 1024;   // memory dim

constexpr int BM = 128, BN = 128, BK = 32, TPB = 256;
constexpr int LDSP = 40;    // padded LDS row stride in bf16 elements (80B, 16B-aligned)

// Branch-free tanh: gfx1250 hardware V_TANH_F32 (co-executes with XDL WMMA).
__device__ __forceinline__ float fast_tanh(float x)
{
#if __has_builtin(__builtin_amdgcn_tanhf)
    return __builtin_amdgcn_tanhf(x);
#elif __has_builtin(__builtin_amdgcn_tanh_f32)
    return __builtin_amdgcn_tanh_f32(x);
#else
    const float ax = __builtin_fabsf(x);
    const float e  = __builtin_amdgcn_exp2f(-2.885390081777927f * ax); // exp(-2|x|)
    const float t  = (1.0f - e) * __builtin_amdgcn_rcpf(1.0f + e);
    return __builtin_copysignf(t, x);
#endif
}

__device__ __forceinline__ void wait_asynccnt0()
{
#if __has_builtin(__builtin_amdgcn_s_wait_asynccnt)
    __builtin_amdgcn_s_wait_asynccnt(0);
#else
    asm volatile("s_wait_asynccnt 0x0" ::: "memory");
#endif
}

// Per-lane 16B global->LDS async copy (GVS mode: saddr base + 32-bit vgpr offset).
__device__ __forceinline__ void async_g2l_b128(const void* gbase, uint32_t voff_bytes,
                                               uint32_t lds_off_bytes)
{
    asm volatile("global_load_async_to_lds_b128 %0, %1, %2"
                 :: "v"(lds_off_bytes), "v"(voff_bytes), "s"(gbase)
                 : "memory");
}

// GEMM with fused predictive-coding epilogues.
//   MODE 0 (INIT):  vals[m,n] = acc
//   MODE 1 (ERR):   err[m,n]  = vals[m,n] - acc            (A gets tanh on load)
//   MODE 2 (DELTA): vals[m,n] += DT*(-err - pen*sign(vals) + (1-tanh(vals)^2)*acc)
// A: fp32 [M x K]; Bw: bf16 [N x K] (i.e. B^T row-major).
template <int MODE, bool ATANH>
__global__ __launch_bounds__(TPB)
void gemm_fused(const float* __restrict__ A, const __bf16* __restrict__ Bw,
                float* __restrict__ vals, float* __restrict__ err,
                int N, int K, float pen)
{
    __shared__ __align__(16) __bf16 sA[2][BM * LDSP];
    __shared__ __align__(16) __bf16 sB[2][BN * LDSP];

    const int tid   = threadIdx.x;
    const int mBase = blockIdx.y * BM;
    const int nBase = blockIdx.x * BN;

    // staging maps (256 threads)
    const int arow = tid >> 3;            // 0..31 (+i*32) -> 128 rows
    const int akq  = (tid & 7) * 4;       // fp32 quad within 32-wide K
    const int brow = tid >> 2;            // 0..63 (+i*64) -> 128 rows
    const int bkq  = (tid & 3) * 8;       // bf16 octet within 32-wide K

    // wave tiling: 8 waves -> 4 (M) x 2 (N); wave tile 32x64 = 2x4 WMMA tiles
    const int lane = tid & 31;
    const int wave = tid >> 5;
    const int lh   = lane & 15;
    const int sel  = lane >> 4;
    const int wmB  = (wave & 3) * 32;
    const int wnB  = (wave >> 2) * 64;

    float4 aReg[4];

    auto loadA = [&](int kt) {
        const int kb = kt * BK;
        #pragma unroll
        for (int i = 0; i < 4; ++i)
            aReg[i] = *(const float4*)(A + (size_t)(mBase + arow + i * 32) * K + kb + akq);
    };
    auto asyncB = [&](int kt, int buf) {
        const int kb = kt * BK;
        #pragma unroll
        for (int i = 0; i < 2; ++i) {
            const uint32_t voff =
                ((uint32_t)(nBase + brow + i * 64) * (uint32_t)K + (uint32_t)(kb + bkq)) * 2u;
            const uint32_t loff =
                (uint32_t)(uintptr_t)&sB[buf][(brow + i * 64) * LDSP + bkq];
            async_g2l_b128(Bw, voff, loff);
        }
    };
    auto storeA = [&](int buf) {
        #pragma unroll
        for (int i = 0; i < 4; ++i) {
            float f4[4] = { aReg[i].x, aReg[i].y, aReg[i].z, aReg[i].w };
            union { __bf16 h[4]; uint2 u; } p;
            #pragma unroll
            for (int j = 0; j < 4; ++j) {
                float f = f4[j];
                if (ATANH) f = fast_tanh(f);
                p.h[j] = (__bf16)f;
            }
            *(uint2*)&sA[buf][(arow + i * 32) * LDSP + akq] = p.u;
        }
    };

    v8f acc[2][4];
    #pragma unroll
    for (int im = 0; im < 2; ++im)
        #pragma unroll
        for (int ii = 0; ii < 4; ++ii)
            #pragma unroll
            for (int v = 0; v < 8; ++v) acc[im][ii][v] = 0.0f;

    const int KT = K / BK;
    asyncB(0, 0);
    loadA(0);
    for (int kt = 0; kt < KT; ++kt) {
        const int buf = kt & 1;
        storeA(buf);              // ds_store A tile kt
        wait_asynccnt0();         // B tile kt resident in LDS[buf]
        __syncthreads();          // publish tile kt (compiler also drains DScnt)
        if (kt + 1 < KT) {        // after barrier: nobody still reads buf^1
            asyncB(kt + 1, buf ^ 1);
            loadA(kt + 1);
        }

        V16U af[2], bfr[4];
        #pragma unroll
        for (int im = 0; im < 2; ++im) {
            // ISA A layout: lanes<16 -> K {0..7,16..23}; lanes>=16 -> {8..15,24..31}
            const __bf16* p = &sA[buf][(wmB + im * 16 + lh) * LDSP + sel * 8];
            af[im].u[0] = *(const uint4*)p;
            af[im].u[1] = *(const uint4*)(p + 16);
        }
        #pragma unroll
        for (int ii = 0; ii < 4; ++ii) {
            // B layout: lane holds column n=lh, K {0..7,16..23} (sel=0) / {8..15,24..31} (sel=1)
            const __bf16* p = &sB[buf][(wnB + ii * 16 + lh) * LDSP + sel * 8];
            bfr[ii].u[0] = *(const uint4*)p;
            bfr[ii].u[1] = *(const uint4*)(p + 16);
        }
        #pragma unroll
        for (int im = 0; im < 2; ++im)
            #pragma unroll
            for (int ii = 0; ii < 4; ++ii)
                acc[im][ii] = __builtin_amdgcn_wmma_f32_16x16x32_bf16(
                    false, af[im].v, false, bfr[ii].v,
                    (short)0, acc[im][ii], false, false);
    }

    // epilogue: C layout -> row = v + 8*sel, col = lh within each 16x16 tile
    #pragma unroll
    for (int im = 0; im < 2; ++im) {
        #pragma unroll
        for (int ii = 0; ii < 4; ++ii) {
            #pragma unroll
            for (int v = 0; v < 8; ++v) {
                const int row = mBase + wmB + im * 16 + sel * 8 + v;
                const int col = nBase + wnB + ii * 16 + lh;
                const size_t idx = (size_t)row * N + col;
                const float a = acc[im][ii][v];
                if (MODE == 0) {
                    vals[idx] = a;
                } else if (MODE == 1) {
                    err[idx] = vals[idx] - a;
                } else {
                    const float vv = vals[idx];
                    const float e  = err[idx];
                    const float th = fast_tanh(vv);
                    const float d  = 1.0f - th * th;
                    const float sg = (vv > 0.0f) ? 1.0f : ((vv < 0.0f) ? -1.0f : 0.0f);
                    vals[idx] = vv + DT_C * (-e - pen * sg + d * a);
                }
            }
        }
    }
}

// Weight convert: Wf = bf16(W) ([out][in]); Wb = bf16(W^T) ([in][out])
__global__ void wconv_kernel(const float* __restrict__ W,
                             __bf16* __restrict__ Wf, __bf16* __restrict__ Wb,
                             int outDim, int inDim)
{
    __shared__ float tile[32][33];
    const int i0 = blockIdx.x * 32;
    const int o0 = blockIdx.y * 32;
    const int tx = threadIdx.x, ty = threadIdx.y;   // 32 x 8
    #pragma unroll
    for (int j = 0; j < 4; ++j) {
        const int o = o0 + ty + j * 8;
        const int i = i0 + tx;
        const float f = W[(size_t)o * inDim + i];
        Wf[(size_t)o * inDim + i] = (__bf16)f;
        tile[ty + j * 8][tx] = f;
    }
    __syncthreads();
    #pragma unroll
    for (int j = 0; j < 4; ++j) {
        const int i = i0 + ty + j * 8;
        const int o = o0 + tx;
        Wb[(size_t)i * outDim + o] = (__bf16)tile[tx][ty + j * 8];
    }
}

__global__ void init0_kernel(float* __restrict__ v0, float* __restrict__ e0,
                             const float* __restrict__ mem, int n)
{
    const int i = blockIdx.x * blockDim.x + threadIdx.x;
    if (i < n) { v0[i] = mem[i & (D0 - 1)]; e0[i] = 0.0f; }
}

__global__ void err0_kernel(float* __restrict__ e0, const float* __restrict__ v0,
                            const float* __restrict__ mem, int n)
{
    const int i = blockIdx.x * blockDim.x + threadIdx.x;
    if (i < n) e0[i] = v0[i] - mem[i & (D0 - 1)];
}

__global__ void zero_kernel(float* __restrict__ p, int n)
{
    const int i = blockIdx.x * blockDim.x + threadIdx.x;
    if (i < n) p[i] = 0.0f;
}

__global__ void copy_kernel(float* __restrict__ d, const float* __restrict__ s, int n)
{
    const int i = blockIdx.x * blockDim.x + threadIdx.x;
    if (i < n) d[i] = s[i];
}

__global__ void sensory_kernel(float* __restrict__ v3, const float* __restrict__ e3,
                               const int* __restrict__ mask, int n)
{
    const int i = blockIdx.x * blockDim.x + threadIdx.x;
    if (i < n) v3[i] = v3[i] - DT_C * e3[i] * (float)mask[i];
}

extern "C" void kernel_launch(void* const* d_in, const int* in_sizes, int n_in,
                              void* d_out, int out_size, void* d_ws, size_t ws_size,
                              hipStream_t stream)
{
    (void)in_sizes; (void)n_in; (void)out_size; (void)ws_size;
    const float* corrupt = (const float*)d_in[0];
    const float* memory  = (const float*)d_in[1];
    const float* W0      = (const float*)d_in[2];
    const float* W1      = (const float*)d_in[3];
    const float* W2      = (const float*)d_in[4];
    const int*   mask    = (const int*)d_in[5];
    float*       out     = (float*)d_out;          // vals[3] lives here
    const int N_ITERS = 20;                        // fixed by setup_inputs()

    // workspace carve-out (256B aligned slices)
    size_t off = 0;
    auto carve = [&](size_t bytes) {
        void* p = (char*)d_ws + off;
        off += (bytes + 255) & ~(size_t)255;
        return p;
    };
    float*  vals0 = (float*)carve((size_t)Bsz * D0 * 4);
    float*  vals1 = (float*)carve((size_t)Bsz * Dm * 4);
    float*  vals2 = (float*)carve((size_t)Bsz * Dm * 4);
    float*  errs0 = (float*)carve((size_t)Bsz * D0 * 4);
    float*  errs1 = (float*)carve((size_t)Bsz * Dm * 4);
    float*  errs2 = (float*)carve((size_t)Bsz * Dm * 4);
    float*  errs3 = (float*)carve((size_t)Bsz * Dm * 4);
    __bf16* Wf0 = (__bf16*)carve((size_t)Dm * D0 * 2);
    __bf16* Wf1 = (__bf16*)carve((size_t)Dm * Dm * 2);
    __bf16* Wf2 = (__bf16*)carve((size_t)Dm * Dm * 2);
    __bf16* Wb0 = (__bf16*)carve((size_t)D0 * Dm * 2);
    __bf16* Wb1 = (__bf16*)carve((size_t)Dm * Dm * 2);
    __bf16* Wb2 = (__bf16*)carve((size_t)Dm * Dm * 2);

    const int nBD  = Bsz * Dm;
    const int nBD0 = Bsz * D0;
    const dim3 blkE(256), blkW(32, 8);
    const dim3 grdBD((nBD + 255) / 256), grdBD0((nBD0 + 255) / 256);

    auto gemm_init = [&](const float* A, const __bf16* Bw, float* vo, int N, int K) {
        dim3 g(N / BN, Bsz / BM);
        gemm_fused<0, true><<<g, TPB, 0, stream>>>(A, Bw, vo, nullptr, N, K, 0.0f);
    };
    auto gemm_err = [&](const float* A, const __bf16* Bw, float* vc, float* eo, int N, int K) {
        dim3 g(N / BN, Bsz / BM);
        gemm_fused<1, true><<<g, TPB, 0, stream>>>(A, Bw, vc, eo, N, K, 0.0f);
    };
    auto gemm_delta = [&](const float* A, const __bf16* Bw, float* vio, float* ei,
                          int N, int K, float pen) {
        dim3 g(N / BN, Bsz / BM);
        gemm_fused<2, false><<<g, TPB, 0, stream>>>(A, Bw, vio, ei, N, K, pen);
    };

    // ---- weight conversion (bf16 L2-resident copies + transposes) ----
    wconv_kernel<<<dim3(D0 / 32, Dm / 32), blkW, 0, stream>>>(W0, Wf0, Wb0, Dm, D0);
    wconv_kernel<<<dim3(Dm / 32, Dm / 32), blkW, 0, stream>>>(W1, Wf1, Wb1, Dm, Dm);
    wconv_kernel<<<dim3(Dm / 32, Dm / 32), blkW, 0, stream>>>(W2, Wf2, Wb2, Dm, Dm);

    // ---- set_nodes + initial errs ----
    init0_kernel<<<grdBD0, blkE, 0, stream>>>(vals0, errs0, memory, nBD0);
    gemm_init(vals0, Wf0, vals1, Dm, D0);          // vals1 = tanh(vals0) @ W0^T
    zero_kernel<<<grdBD, blkE, 0, stream>>>(errs1, nBD);
    gemm_init(vals1, Wf1, vals2, Dm, Dm);          // vals2 = tanh(vals1) @ W1^T
    zero_kernel<<<grdBD, blkE, 0, stream>>>(errs2, nBD);
    copy_kernel<<<grdBD, blkE, 0, stream>>>(out, corrupt, nBD);   // vals3 = corrupt
    gemm_err(vals2, Wf2, out, errs3, Dm, Dm);      // errs3 = vals3 - tanh(vals2) @ W2^T

    // ---- inference iterations ----
    for (int it = 0; it < N_ITERS; ++it) {
        // deltas (consume old errs, update vals in place)
        gemm_delta(errs1, Wb0, vals0, errs0, D0, Dm, LAMB_C);  // G0 = errs1 @ W0
        gemm_delta(errs2, Wb1, vals1, errs1, Dm, Dm, 0.0f);    // G1 = errs2 @ W1
        gemm_delta(errs3, Wb2, vals2, errs2, Dm, Dm, 0.0f);    // G2 = errs3 @ W2
        sensory_kernel<<<grdBD, blkE, 0, stream>>>(out, errs3, mask, nBD);
        // recompute errs from new vals
        err0_kernel<<<grdBD0, blkE, 0, stream>>>(errs0, vals0, memory, nBD0);
        gemm_err(vals0, Wf0, vals1, errs1, Dm, D0);
        gemm_err(vals1, Wf1, vals2, errs2, Dm, Dm);
        gemm_err(vals2, Wf2, out,   errs3, Dm, Dm);
    }
    // out already holds vals[3]
}
